// SpatialProximityHead_14302241096162
// MI455X (gfx1250) — matrix operand order repaired and checked by reading, not compiled, not run on hardware
//
#include <hip/hip_runtime.h>

constexpr int kH  = 256;
constexpr int kNH = 4;
constexpr int kDK = 64;
constexpr int kNL = 2;
constexpr int kTL = 6;
constexpr int kN  = 512;

typedef _Float16 v16h __attribute__((ext_vector_type(16)));
typedef _Float16 v8h  __attribute__((ext_vector_type(8)));
typedef float    v8f  __attribute__((ext_vector_type(8)));

// ---------------------------------------------------------------------------
// f32 -> f16 conversion
// ---------------------------------------------------------------------------
__global__ void f32_to_f16(const float* __restrict__ src, _Float16* __restrict__ dst, int n) {
    int i = blockIdx.x * blockDim.x + threadIdx.x;
    if (i < n) dst[i] = (_Float16)src[i];
}

// f32 (Kdim x Ndim row-major) -> f16 transposed (Ndim x Kdim row-major).
// All GEMM B operands become K-contiguous -> 32B vector fragment loads.
__global__ void f32_to_f16_T(const float* __restrict__ src, _Float16* __restrict__ dst,
                             int Kdim, int Ndim) {
    int idx = blockIdx.x * blockDim.x + threadIdx.x;
    if (idx < Kdim * Ndim) {
        const int k = idx / Ndim, n = idx % Ndim;
        dst[(long long)n * Kdim + k] = (_Float16)src[idx];
    }
}

// ---------------------------------------------------------------------------
// WMMA GEMM:  C[m,n] = act( sum_k A[m,k] * Bt[n,k] + bias[n] )
//   A : f16 row-major (M x K), lda, optional per-head stride
//   Bt: f16 row-major (N x K)  (B transposed; K-contiguous)
//   One wave = 16x64 macro-tile: 4 accumulators share each A fragment
//   (4x WMMA per A load), 2-stage software pipeline on all fragments.
//   Outputs: f32 C and/or f16 C16; TR16 stores C16 transposed (n-major).
// ---------------------------------------------------------------------------
template <bool RELU, bool HAS_C, bool HAS_C16, bool TR16>
__global__ void wmma_gemm(const _Float16* __restrict__ A, int lda, long long aHead,
                          const _Float16* __restrict__ B, int ldb, long long bHead,
                          const float* __restrict__ bias,
                          float* __restrict__ C, int ldc, long long cHead,
                          _Float16* __restrict__ C16, int ldc16, long long c16Head,
                          int M, int Nn, int K) {
    const int lane   = threadIdx.x & 31;
    const int wave   = blockIdx.x * (blockDim.x >> 5) + (threadIdx.x >> 5);
    const int tilesN = Nn >> 6;                      // 64-wide macro tiles
    const int tiles  = (M >> 4) * tilesN;
    if (wave >= tiles) return;                       // wave-uniform; EXEC stays full
    const int head = blockIdx.y;

    const int tm = wave / tilesN, tn = wave % tilesN;
    const int m0 = tm << 4, n0 = tn << 6;
    const int hi = lane >> 4;                        // lane half: 0/1
    const int lx = lane & 15;

    const _Float16* arow = A + (long long)head * aHead + (long long)(m0 + lx) * lda;
    const _Float16* brow = B + (long long)head * bHead + (long long)(n0 + lx) * ldb;

    v8f acc[4];
#pragma unroll
    for (int t = 0; t < 4; ++t) {
        const float bv = bias ? bias[n0 + (t << 4) + lx] : 0.0f;   // bias: n only
#pragma unroll
        for (int v = 0; v < 8; ++v) acc[t][v] = bv;
    }

    // Fragment loaders (this code's contiguous k order, same for A and B):
    //  A: lane lx = row; regs hold K = hi*16 + {0..15} (contiguous, matches B)
    //  B: lane lx = col; regs hold K = hi*16 + {0..15} (contiguous 32B)
    auto loadA = [&](int k0) -> v16h {
        v8h p0 = *(const v8h*)(arow + k0 + (hi << 4));
        v8h p1 = *(const v8h*)(arow + k0 + (hi << 4) + 8);
        return __builtin_shufflevector(p0, p1, 0, 1, 2, 3, 4, 5, 6, 7,
                                       8, 9, 10, 11, 12, 13, 14, 15);
    };
    auto loadB = [&](int t, int k0) -> v16h {
        return *(const v16h*)(brow + (long long)(t << 4) * ldb + k0 + (hi << 4));
    };

    v16h a0  = loadA(0);
    v16h b00 = loadB(0, 0), b01 = loadB(1, 0), b02 = loadB(2, 0), b03 = loadB(3, 0);
#pragma unroll 2
    for (int k0 = 32; k0 < K; k0 += 32) {
        v16h a1  = loadA(k0);
        v16h b10 = loadB(0, k0), b11 = loadB(1, k0), b12 = loadB(2, k0), b13 = loadB(3, k0);
        acc[0] = __builtin_amdgcn_wmma_f32_16x16x32_f16(false, a0, false, b00, (short)0, acc[0], false, false);
        acc[1] = __builtin_amdgcn_wmma_f32_16x16x32_f16(false, a0, false, b01, (short)0, acc[1], false, false);
        acc[2] = __builtin_amdgcn_wmma_f32_16x16x32_f16(false, a0, false, b02, (short)0, acc[2], false, false);
        acc[3] = __builtin_amdgcn_wmma_f32_16x16x32_f16(false, a0, false, b03, (short)0, acc[3], false, false);
        a0 = a1; b00 = b10; b01 = b11; b02 = b12; b03 = b13;
    }
    acc[0] = __builtin_amdgcn_wmma_f32_16x16x32_f16(false, a0, false, b00, (short)0, acc[0], false, false);
    acc[1] = __builtin_amdgcn_wmma_f32_16x16x32_f16(false, a0, false, b01, (short)0, acc[1], false, false);
    acc[2] = __builtin_amdgcn_wmma_f32_16x16x32_f16(false, a0, false, b02, (short)0, acc[2], false, false);
    acc[3] = __builtin_amdgcn_wmma_f32_16x16x32_f16(false, a0, false, b03, (short)0, acc[3], false, false);

#pragma unroll
    for (int t = 0; t < 4; ++t) {
#pragma unroll
        for (int v = 0; v < 8; ++v) {
            const int m = m0 + (hi << 3) + v;
            const int n = n0 + (t << 4) + lx;
            float val = acc[t][v];
            if (RELU) val = fmaxf(val, 0.0f);
            if (HAS_C)
                C[(long long)head * cHead + (long long)m * ldc + n] = val;
            if (HAS_C16) {
                if (TR16)
                    C16[(long long)head * c16Head + (long long)n * ldc16 + m] = (_Float16)val;
                else
                    C16[(long long)head * c16Head + (long long)m * ldc16 + n] = (_Float16)val;
            }
        }
    }
}

// ---------------------------------------------------------------------------
// dist / dw. Reference normalizes dw over the QUERY axis (axis=2 of
// (B,1,Nq,Nk)) -> per key column j. One block per column j.
// ---------------------------------------------------------------------------
__global__ void dist_kernel(const float* __restrict__ preds,
                            float* __restrict__ dw, float* __restrict__ dist) {
    __shared__ float red[256];
    const float* base = preds + (size_t)(kTL - 1) * kN * 33;
    const int j = blockIdx.x;
    const float sx = base[j * 33 + 0], sy = base[j * 33 + 1], sz = base[j * 33 + 2];
    float dv[2]; float local = 0.0f;
#pragma unroll
    for (int t = 0; t < 2; ++t) {
        const int i = threadIdx.x + (t << 8);
        const float dx = sx - base[i * 33 + 30];
        const float dy = sy - base[i * 33 + 31];
        const float dz = sz - base[i * 33 + 32];
        const float d = sqrtf(dx * dx + dy * dy + dz * dz);
        dv[t] = d;
        dist[(long long)i * kN + j] = d;
        local += 1.0f / (d + 0.01f);
    }
    red[threadIdx.x] = local; __syncthreads();
    for (int s = 128; s > 0; s >>= 1) {
        if (threadIdx.x < s) red[threadIdx.x] += red[threadIdx.x + s];
        __syncthreads();
    }
    const float inv = 1.0f / red[0];
#pragma unroll
    for (int t = 0; t < 2; ++t) {
        const int i = threadIdx.x + (t << 8);
        dw[(long long)i * kN + j] = (1.0f / (dv[t] + 0.01f)) * inv;
    }
}

// ---------------------------------------------------------------------------
// softmax over keys, per-head bias: h0 -> +dw, h1 -> -dist, h2/3 -> 0.
// ---------------------------------------------------------------------------
__global__ void softmax_bias(const float* __restrict__ scores,
                             const float* __restrict__ dw,
                             const float* __restrict__ dist,
                             _Float16* __restrict__ att) {
    __shared__ float red[256];
    const int h = blockIdx.x >> 9;
    const int i = blockIdx.x & 511;
    const long long rowoff = ((long long)h * kN + i) * kN;
    float vals[2];
#pragma unroll
    for (int t = 0; t < 2; ++t) {
        const int j = threadIdx.x + (t << 8);
        float bterm = 0.0f;
        if (h == 0)      bterm =  dw[(long long)i * kN + j];
        else if (h == 1) bterm = -dist[(long long)i * kN + j];
        vals[t] = scores[rowoff + j] * 0.125f + bterm;   // 1/sqrt(64)
    }
    float m = fmaxf(vals[0], vals[1]);
    red[threadIdx.x] = m; __syncthreads();
    for (int s = 128; s > 0; s >>= 1) {
        if (threadIdx.x < s) red[threadIdx.x] = fmaxf(red[threadIdx.x], red[threadIdx.x + s]);
        __syncthreads();
    }
    m = red[0]; __syncthreads();
    const float e0 = expf(vals[0] - m), e1 = expf(vals[1] - m);
    red[threadIdx.x] = e0 + e1; __syncthreads();
    for (int s = 128; s > 0; s >>= 1) {
        if (threadIdx.x < s) red[threadIdx.x] += red[threadIdx.x + s];
        __syncthreads();
    }
    const float inv = 1.0f / red[0];
    att[rowoff + threadIdx.x]       = (_Float16)(e0 * inv);
    att[rowoff + threadIdx.x + 256] = (_Float16)(e1 * inv);
}

// ---------------------------------------------------------------------------
// x = LayerNorm(x + xo) * g + b ; writes f32 (residual) + f16 (next GEMM A).
// ---------------------------------------------------------------------------
__global__ void add_ln(float* __restrict__ x, const float* __restrict__ xo,
                       const float* __restrict__ g, const float* __restrict__ b,
                       _Float16* __restrict__ x16) {
    __shared__ float red[256];
    const int i = blockIdx.x, c = threadIdx.x;
    const long long idx = (long long)i * kH + c;
    const float v = x[idx] + xo[idx];
    red[c] = v; __syncthreads();
    for (int s = 128; s > 0; s >>= 1) { if (c < s) red[c] += red[c + s]; __syncthreads(); }
    const float mean = red[0] * (1.0f / kH);
    __syncthreads();
    const float d = v - mean;
    red[c] = d * d; __syncthreads();
    for (int s = 128; s > 0; s >>= 1) { if (c < s) red[c] += red[c + s]; __syncthreads(); }
    const float var = red[0] * (1.0f / kH);
    const float y = d * rsqrtf(var + 1e-5f) * g[c] + b[c];
    x[idx] = y;
    x16[idx] = (_Float16)y;
}

// ---------------------------------------------------------------------------
// Pairwise head: lclc[i,j] = b2 + sum_c relu(A[i,c]+Bm[j,c]) * W2[c]
// ---------------------------------------------------------------------------
constexpr int kSP = kH + 8;

__global__ void pair_head(const _Float16* __restrict__ A16, const _Float16* __restrict__ B16,
                          const float* __restrict__ W2, const float* __restrict__ b2p,
                          float* __restrict__ out) {
    __shared__ _Float16 As[32 * kSP];
    __shared__ _Float16 Bs[32 * kSP];
    __shared__ float w2s[kH];
    const int i0 = blockIdx.y << 5, j0 = blockIdx.x << 5;
    for (int t = threadIdx.x; t < 32 * kH; t += 256) {
        const int r = t >> 8, c = t & 255;
        As[r * kSP + c] = A16[(long long)(i0 + r) * kH + c];
        Bs[r * kSP + c] = B16[(long long)(j0 + r) * kH + c];
    }
    w2s[threadIdx.x] = W2[threadIdx.x];
    __syncthreads();
    const float b2 = b2p[0];
    const int jj  = threadIdx.x & 31;
    const int ii0 = threadIdx.x >> 5;
    for (int r = 0; r < 4; ++r) {
        const int ii = ii0 + (r << 3);
        const _Float16* ap = &As[ii * kSP];
        const _Float16* bp = &Bs[jj * kSP];
        float acc = b2;
#pragma unroll 8
        for (int c = 0; c < kH; ++c)
            acc += fmaxf((float)ap[c] + (float)bp[c], 0.0f) * w2s[c];
        const long long idx = (long long)(i0 + ii) * kN + (j0 + jj);
#pragma unroll
        for (int t = 0; t < kTL; ++t)
            out[(long long)t * kN * kN + idx] = acc;
    }
}

// ---------------------------------------------------------------------------
// Host side
// ---------------------------------------------------------------------------
static inline dim3 gemm_grid(int M, int Nn, int heads) {
    return dim3((unsigned)(((M / 16) * (Nn / 64) + 7) / 8), (unsigned)heads);
}

extern "C" void kernel_launch(void* const* d_in, const int* in_sizes, int n_in,
                              void* d_out, int out_size, void* d_ws, size_t ws_size,
                              hipStream_t stream) {
    (void)in_sizes; (void)n_in; (void)out_size; (void)ws_size;
    const float* history = (const float*)d_in[0];
    const float* preds   = (const float*)d_in[1];
    const float* fc_W    = (const float*)d_in[2];
    const float* fc_b    = (const float*)d_in[3];
    const float* Wq      = (const float*)d_in[4];
    const float* bq      = (const float*)d_in[5];
    const float* Wk      = (const float*)d_in[6];
    const float* bk      = (const float*)d_in[7];
    const float* Wv      = (const float*)d_in[8];
    const float* bv      = (const float*)d_in[9];
    const float* Wo      = (const float*)d_in[10];
    const float* bo      = (const float*)d_in[11];
    const float* ln_g    = (const float*)d_in[12];
    const float* ln_b    = (const float*)d_in[13];
    const float* mlp1_W  = (const float*)d_in[14];
    const float* mlp1_b  = (const float*)d_in[15];
    const float* mlp2_W  = (const float*)d_in[16];
    const float* mlp2_b  = (const float*)d_in[17];
    const float* cls_W1  = (const float*)d_in[18];
    const float* cls_b1  = (const float*)d_in[19];
    const float* cls_W2  = (const float*)d_in[20];
    const float* cls_b2  = (const float*)d_in[21];
    float* out = (float*)d_out;

    char* wsb = (char*)d_ws;
    size_t off = 0;
    auto alloc = [&](size_t bytes) -> void* {
        void* p = wsb + off;
        off += (bytes + 255) & ~(size_t)255;
        return p;
    };
    _Float16* hs16  = (_Float16*)alloc((size_t)kN * kH * 2);
    _Float16* wfcT  = (_Float16*)alloc((size_t)kH * kH * 2);
    _Float16* wqT   = (_Float16*)alloc((size_t)kNL * kH * kH * 2);
    _Float16* wkT   = (_Float16*)alloc((size_t)kNL * kH * kH * 2);
    _Float16* wvT   = (_Float16*)alloc((size_t)kNL * kH * kH * 2);
    _Float16* woT   = (_Float16*)alloc((size_t)kNL * kH * kH * 2);
    _Float16* wm1T  = (_Float16*)alloc((size_t)kH * kH * 2);
    _Float16* wm2T  = (_Float16*)alloc((size_t)kH * kH * 2);
    _Float16* w1aT  = (_Float16*)alloc((size_t)kH * kH * 2);
    _Float16* w1bT  = (_Float16*)alloc((size_t)kH * kH * 2);
    float*    xbuf  = (float*)alloc((size_t)kN * kH * 4);
    _Float16* x16   = (_Float16*)alloc((size_t)kN * kH * 2);
    _Float16* q16   = (_Float16*)alloc((size_t)kN * kH * 2);
    _Float16* k16   = (_Float16*)alloc((size_t)kN * kH * 2);
    _Float16* vT16  = (_Float16*)alloc((size_t)kH * kN * 2);   // V stored transposed
    float*    dwb   = (float*)alloc((size_t)kN * kN * 4);
    float*    distb = (float*)alloc((size_t)kN * kN * 4);
    float*    scores= (float*)alloc((size_t)kNH * kN * kN * 4);
    _Float16* att16 = (_Float16*)alloc((size_t)kNH * kN * kN * 2);
    _Float16* ao16  = (_Float16*)alloc((size_t)kN * kH * 2);
    float*    xo    = (float*)alloc((size_t)kN * kH * 4);
    _Float16* o1    = (_Float16*)alloc((size_t)kN * kH * 2);
    _Float16* o2    = (_Float16*)alloc((size_t)kN * kH * 2);
    _Float16* A16   = (_Float16*)alloc((size_t)kN * kH * 2);
    _Float16* B16m  = (_Float16*)alloc((size_t)kN * kH * 2);

    auto conv = [&](const float* s, _Float16* d, int n) {
        f32_to_f16<<<(n + 255) / 256, 256, 0, stream>>>(s, d, n);
    };
    auto convT = [&](const float* s, _Float16* d, int Kd, int Nd) {
        f32_to_f16_T<<<(Kd * Nd + 255) / 256, 256, 0, stream>>>(s, d, Kd, Nd);
    };
    conv(history + (size_t)(kTL - 1) * kN * kH, hs16, kN * kH);
    convT(fc_W, wfcT, kH, kH);
    for (int l = 0; l < kNL; ++l) {
        const size_t w = (size_t)l * kH * kH;
        convT(Wq + w, wqT + w, kH, kH);
        convT(Wk + w, wkT + w, kH, kH);
        convT(Wv + w, wvT + w, kH, kH);
        convT(Wo + w, woT + w, kH, kH);
    }
    convT(mlp1_W, wm1T, kH, kH);
    convT(mlp2_W, wm2T, kH, kH);
    convT(cls_W1, w1aT, kH, kH);                       // rows 0..H-1  of cls_W1
    convT(cls_W1 + (size_t)kH * kH, w1bT, kH, kH);     // rows H..2H-1 of cls_W1

    dist_kernel<<<kN, 256, 0, stream>>>(preds, dwb, distb);

    // x = relu(qf @ fc_W + fc_b)  -> f32 + f16
    wmma_gemm<true, true, true, false><<<gemm_grid(kN, kH, 1), 256, 0, stream>>>(
        hs16, kH, 0, wfcT, kH, 0, fc_b, xbuf, kH, 0, x16, kH, 0, kN, kH, kH);

    for (int l = 0; l < kNL; ++l) {
        const long long wOff = (long long)l * kH * kH;
        // Q, K projections (f16); V projection stored transposed (kH x kN)
        wmma_gemm<false, false, true, false><<<gemm_grid(kN, kH, 1), 256, 0, stream>>>(
            x16, kH, 0, wqT + wOff, kH, 0, bq + l * kH,
            nullptr, 0, 0, q16, kH, 0, kN, kH, kH);
        wmma_gemm<false, false, true, false><<<gemm_grid(kN, kH, 1), 256, 0, stream>>>(
            x16, kH, 0, wkT + wOff, kH, 0, bk + l * kH,
            nullptr, 0, 0, k16, kH, 0, kN, kH, kH);
        wmma_gemm<false, false, true, true><<<gemm_grid(kN, kH, 1), 256, 0, stream>>>(
            x16, kH, 0, wvT + wOff, kH, 0, bv + l * kH,
            nullptr, 0, 0, vT16, kN, 0, kN, kH, kH);
        // scores[h] = q_h @ k_h^T   (k16 rows are K-contiguous already)
        wmma_gemm<false, true, false, false><<<gemm_grid(kN, kN, kNH), 256, 0, stream>>>(
            q16, kH, kDK, k16, kH, kDK, nullptr,
            scores, kN, (long long)kN * kN, nullptr, 0, 0, kN, kN, kDK);
        softmax_bias<<<kNH * kN, 256, 0, stream>>>(scores, dwb, distb, att16);
        // attout[h] = att_h @ v_h   (B = vT16: rows d, K-contiguous over tokens)
        wmma_gemm<false, false, true, false><<<gemm_grid(kN, kDK, kNH), 256, 0, stream>>>(
            att16, kN, (long long)kN * kN, vT16, kN, (long long)kDK * kN, nullptr,
            nullptr, 0, 0, ao16, kH, kDK, kN, kDK, kN);
        // out proj (f32) then residual + LN
        wmma_gemm<false, true, false, false><<<gemm_grid(kN, kH, 1), 256, 0, stream>>>(
            ao16, kH, 0, woT + wOff, kH, 0, bo + l * kH,
            xo, kH, 0, nullptr, 0, 0, kN, kH, kH);
        add_ln<<<kN, 256, 0, stream>>>(xbuf, xo, ln_g + l * kH, ln_b + l * kH, x16);
    }

    // o1 = relu(x@mlp1+b), o2 = relu(x@mlp2+b)
    wmma_gemm<true, false, true, false><<<gemm_grid(kN, kH, 1), 256, 0, stream>>>(
        x16, kH, 0, wm1T, kH, 0, mlp1_b, nullptr, 0, 0, o1, kH, 0, kN, kH, kH);
    wmma_gemm<true, false, true, false><<<gemm_grid(kN, kH, 1), 256, 0, stream>>>(
        x16, kH, 0, wm2T, kH, 0, mlp2_b, nullptr, 0, 0, o2, kH, 0, kN, kH, kH);
    // A = o1 @ W1a + b1 ; Bm = o2 @ W1b   (separable pairwise classifier)
    wmma_gemm<false, false, true, false><<<gemm_grid(kN, kH, 1), 256, 0, stream>>>(
        o1, kH, 0, w1aT, kH, 0, cls_b1, nullptr, 0, 0, A16, kH, 0, kN, kH, kH);
    wmma_gemm<false, false, true, false><<<gemm_grid(kN, kH, 1), 256, 0, stream>>>(
        o2, kH, 0, w1bT, kH, 0, nullptr, nullptr, 0, 0, B16m, kH, 0, kN, kH, kH);

    dim3 pg(kN / 32, kN / 32);
    pair_head<<<pg, 256, 0, stream>>>(A16, B16m, cls_W2, cls_b2, out);
}
